// BERT_77008763617386
// MI455X (gfx1250) — compile-verified, matrix operand order checked by reading, not run on hardware
//
#include <hip/hip_runtime.h>
#include <cstdint>
#include <cstddef>

// ---------------- problem constants ----------------
#define B_   8
#define S_   512
#define V_   30522
#define P_   10000
#define H_   4
#define DI_  512
#define DO_  128
#define NTOK (B_ * S_)   // 4096
#define F_   (5 * DI_)   // 2560

// ---------------- WMMA types ----------------
typedef __attribute__((ext_vector_type(16))) __bf16 v16bf;
typedef __attribute__((ext_vector_type(8)))  float  v8f;

union BF16Frag { v16bf v; unsigned int w[8]; unsigned short u[16]; };

__device__ __forceinline__ unsigned short f32_to_bf16(float f) {
  unsigned int x = __float_as_uint(f);
  x += 0x7FFFu + ((x >> 16) & 1u);   // round to nearest even
  return (unsigned short)(x >> 16);
}

// ---------------- block reductions (wave32) ----------------
__device__ __forceinline__ float block_reduce_sum(float v, float* sred) {
#pragma unroll
  for (int off = 16; off > 0; off >>= 1) v += __shfl_xor(v, off, 32);
  const int w = threadIdx.x >> 5;
  if ((threadIdx.x & 31) == 0) sred[w] = v;
  __syncthreads();
  const int nw = (blockDim.x + 31) >> 5;
  if (threadIdx.x < 32) {
    float x = (threadIdx.x < nw) ? sred[threadIdx.x] : 0.0f;
#pragma unroll
    for (int off = 16; off > 0; off >>= 1) x += __shfl_xor(x, off, 32);
    if (threadIdx.x == 0) sred[0] = x;
  }
  __syncthreads();
  float r = sred[0];
  __syncthreads();
  return r;
}

__device__ __forceinline__ float block_reduce_max(float v, float* sred) {
#pragma unroll
  for (int off = 16; off > 0; off >>= 1) v = fmaxf(v, __shfl_xor(v, off, 32));
  const int w = threadIdx.x >> 5;
  if ((threadIdx.x & 31) == 0) sred[w] = v;
  __syncthreads();
  const int nw = (blockDim.x + 31) >> 5;
  if (threadIdx.x < 32) {
    float x = (threadIdx.x < nw) ? sred[threadIdx.x] : -3.0e38f;
#pragma unroll
    for (int off = 16; off > 0; off >>= 1) x = fmaxf(x, __shfl_xor(x, off, 32));
    if (threadIdx.x == 0) sred[0] = x;
  }
  __syncthreads();
  float r = sred[0];
  __syncthreads();
  return r;
}

// ---------------- fp32 [K][N] -> bf16 [N][K] transpose-convert ----------------
// One matrix per blockIdx.z (contiguous batch). K % 32 == 0; N may be ragged.
__global__ __launch_bounds__(256)
void cvt_transpose_bf16_kernel(const float* __restrict__ src,
                               unsigned short* __restrict__ dst, int K, int N) {
  __shared__ float tile[32][33];
  const float* s = src + (size_t)blockIdx.z * K * N;
  unsigned short* d = dst + (size_t)blockIdx.z * K * N;
  const int k0 = blockIdx.x * 32, n0 = blockIdx.y * 32;
  const int tx = threadIdx.x & 31, ty = threadIdx.x >> 5;   // 32 x 8
#pragma unroll
  for (int i = 0; i < 32; i += 8) {
    int n = n0 + tx, k = k0 + ty + i;
    tile[ty + i][tx] = (n < N) ? s[(size_t)k * N + n] : 0.0f;
  }
  __syncthreads();
#pragma unroll
  for (int i = 0; i < 32; i += 8) {
    int n = n0 + ty + i, k = k0 + tx;
    if (n < N) d[(size_t)n * K + k] = f32_to_bf16(tile[tx][ty + i]);
  }
}

// ---------------- embedding + LN + concat builder ----------------
// cat row layout: [pure | pos_n | te_n | semb_n | poi_n], each 512 cols, bf16.
__global__ __launch_bounds__(256)
void build_cat_kernel(const int* __restrict__ tok, const int* __restrict__ poiI,
                      const float* __restrict__ timeT,
                      const float* __restrict__ s_emb, const float* __restrict__ spat,
                      const float* __restrict__ poitab,
                      const float* __restrict__ w_time, const float* __restrict__ b_time,
                      unsigned short* __restrict__ cat) {
  __shared__ float sred[8];
  const int row = blockIdx.x;            // token index 0..4095
  const int s   = row & (S_ - 1);
  const int t   = tok[row];
  const int p   = poiI[row];
  const float tv = timeT[row];
  unsigned short* crow = cat + (size_t)row * F_;
  const int c0 = threadIdx.x, c1 = threadIdx.x + 256;

  auto lnStore = [&](float x0, float x1, unsigned short* dst) {
    float sum = block_reduce_sum(x0 + x1, sred);
    float sq  = block_reduce_sum(x0 * x0 + x1 * x1, sred);
    float mean = sum * (1.0f / (float)DI_);
    float var  = sq  * (1.0f / (float)DI_) - mean * mean;
    float inv  = rsqrtf(var + 1e-5f);
    dst[c0] = f32_to_bf16((x0 - mean) * inv);
    dst[c1] = f32_to_bf16((x1 - mean) * inv);
  };

  // seg0: pure = LN(spatial_table[tok])
  lnStore(spat[(size_t)t * DI_ + c0], spat[(size_t)t * DI_ + c1], crow);
  // seg1: pos_n = LN(pe[s]); angle = s / 10000^(2c/512); even cols sin, odd cos
  const float kLn = 9.210340371976184f * 2.0f / (float)DI_;
  float a0 = (float)s * __expf(-kLn * (float)c0);
  float a1 = (float)s * __expf(-kLn * (float)c1);
  float p0 = (c0 & 1) ? cosf(a0) : sinf(a0);
  float p1 = (c1 & 1) ? cosf(a1) : sinf(a1);
  lnStore(p0, p1, crow + DI_);
  // seg2: te_n = LN(cos(t*w + b) * sqrt(1/D))
  const float sc = 0.044194173824159216f;   // sqrt(1/512)
  float te0 = cosf(tv * w_time[c0] + b_time[c0]) * sc;
  float te1 = cosf(tv * w_time[c1] + b_time[c1]) * sc;
  lnStore(te0, te1, crow + 2 * DI_);
  // seg3: semb_n = LN(s_emb_table[tok])
  lnStore(s_emb[(size_t)t * DI_ + c0], s_emb[(size_t)t * DI_ + c1], crow + 3 * DI_);
  // seg4: poi_n = LN(poi_table[poi])
  lnStore(poitab[(size_t)p * DI_ + c0], poitab[(size_t)p * DI_ + c1], crow + 4 * DI_);
}

// ---------------- batched bf16 WMMA GEMM ----------------
// C[z] = act( alpha * A[z](MxK, row-major) * B[z] + bias[z] )
// B is stored N-major: logical B[k][n] = Bglob[n*ldb + k]  (all weights pre-transposed)
// Requirements: M % 128 == 0, K % 32 == 0 (all uses satisfy this).
// Template: NCHECK — ragged N (vocab head); otherwise N % 128 == 0
//           FLAGS  — bit0 store bf16 (else f32), bit1 exact-erf GELU,
//                    bit2 has bias, bit3 store C transposed (C[n][m], ldc = row stride of n)
#define GBM 128
#define GBN 128
#define GBK 32
#define LDSS 40   // padded LDS row stride (bf16): 80B rows -> conflict-free, 16B aligned

template <int NCHECK, int FLAGS>
__global__ __launch_bounds__(256, 2)
void gemm_bf16_kernel(const unsigned short* __restrict__ A, int lda, long long Ao, long long Ai,
                      const unsigned short* __restrict__ Bm, int ldb, long long Bo, long long Bi,
                      const float* __restrict__ bias, long long bo, long long bi,
                      void* __restrict__ Cv, int ldc, long long Co, long long Ci,
                      int M, int N, int K, int zdiv, float alpha) {
  __shared__ unsigned short As[GBM * LDSS];   // As[m][k]
  __shared__ unsigned short Bs[GBN * LDSS];   // Bs[n][k]

  const int z  = blockIdx.z;
  const int zq = z / zdiv, zr = z % zdiv;
  const unsigned short* Ab = A  + (size_t)(zq * Ao + zr * Ai);
  const unsigned short* Bb = Bm + (size_t)(zq * Bo + zr * Bi);
  const float* biasb = (FLAGS & 4) ? bias + (size_t)(zq * bo + zr * bi) : nullptr;
  const size_t coff = (size_t)(zq * Co + zr * Ci);

  const int blockM = blockIdx.y * GBM;
  const int blockN = blockIdx.x * GBN;

  const int tid  = threadIdx.x;
  const int lane = tid & 31;
  const int wave = tid >> 5;
  const int wm   = wave & 3;    // 4 waves along M -> 32 rows each
  const int wn   = wave >> 2;   // 2 waves along N -> 64 cols each
  const int half = lane >> 4;
  const int l16  = lane & 15;

  v8f acc[2][4];
  const v8f vzero = {0.f, 0.f, 0.f, 0.f, 0.f, 0.f, 0.f, 0.f};
#pragma unroll
  for (int tm = 0; tm < 2; ++tm)
#pragma unroll
    for (int tn = 0; tn < 4; ++tn) acc[tm][tn] = vzero;

  // global->LDS staging assignments (256 threads; both tiles are K-contiguous)
  const int mA = tid >> 1;            // 0..127
  const int kA = (tid & 1) << 4;      // 0 or 16
  const int nB = tid >> 1;            // 0..127
  const int kC = (tid & 1) << 4;      // 0 or 16
  const bool bRowOK = (NCHECK == 0) || (blockN + nB < N);

  for (int k0 = 0; k0 < K; k0 += GBK) {
    // --- stage A tile ---
    {
      const unsigned short* g = Ab + (size_t)(blockM + mA) * lda + (k0 + kA);
      uint4 x0 = *(const uint4*)g;
      uint4 x1 = *(const uint4*)(g + 8);
      *(uint4*)&As[mA * LDSS + kA]     = x0;
      *(uint4*)&As[mA * LDSS + kA + 8] = x1;
      if (k0 + GBK < K) __builtin_prefetch(g + GBK, 0, 1);
    }
    // --- stage B tile (already N-major in memory) ---
    {
      const unsigned short* g = Bb + (size_t)(blockN + nB) * ldb + (k0 + kC);
      uint4 x0 = {0u, 0u, 0u, 0u}, x1 = {0u, 0u, 0u, 0u};
      if (bRowOK) {
        x0 = *(const uint4*)g;
        x1 = *(const uint4*)(g + 8);
        if (k0 + GBK < K) __builtin_prefetch(g + GBK, 0, 1);
      }
      *(uint4*)&Bs[nB * LDSS + kC]     = x0;
      *(uint4*)&Bs[nB * LDSS + kC + 8] = x1;
    }
    __syncthreads();

    // --- gather fragments per documented 16-bit A/B layouts (wave32) ---
    BF16Frag afr[2], bfr[4];
#pragma unroll
    for (int tm = 0; tm < 2; ++tm) {
      const unsigned short* Ar = &As[(wm * 32 + tm * 16 + l16) * LDSS];
#pragma unroll
      for (int j = 0; j < 8; ++j) {
        int base = ((j < 4) ? 0 : 16) + 8 * half + ((j & 3) << 1);
        afr[tm].w[j] = *(const unsigned int*)(Ar + base);
      }
    }
#pragma unroll
    for (int tn = 0; tn < 4; ++tn) {
      const unsigned short* Br = &Bs[(wn * 64 + tn * 16 + l16) * LDSS];
#pragma unroll
      for (int j = 0; j < 8; ++j) {
        int base = ((j < 4) ? 0 : 16) + 8 * half + ((j & 3) << 1);
        bfr[tn].w[j] = *(const unsigned int*)(Br + base);
      }
    }

#pragma unroll
    for (int tm = 0; tm < 2; ++tm)
#pragma unroll
      for (int tn = 0; tn < 4; ++tn)
        acc[tm][tn] = __builtin_amdgcn_wmma_f32_16x16x32_bf16(
            false, afr[tm].v, false, bfr[tn].v, (short)0, acc[tm][tn], false, false);

    __syncthreads();
  }

  // --- epilogue: C layout lane=(col l16, half), VGPR i = row (i + 8*half) ---
#pragma unroll
  for (int tm = 0; tm < 2; ++tm) {
#pragma unroll
    for (int tn = 0; tn < 4; ++tn) {
      const int col = blockN + wn * 64 + tn * 16 + l16;
      const bool ok = (NCHECK == 0) || (col < N);
      float bv = 0.0f;
      if ((FLAGS & 4) && ok) bv = biasb[col];
      v8f c = acc[tm][tn];
#pragma unroll
      for (int i = 0; i < 8; ++i) {
        int rowg = blockM + wm * 32 + tm * 16 + 8 * half + i;
        float v = c[i] * alpha + bv;
        if (FLAGS & 2) v = 0.5f * v * (1.0f + erff(v * 0.7071067811865475f));
        if (ok) {
          size_t idx = (FLAGS & 8) ? coff + (size_t)col * ldc + rowg
                                   : coff + (size_t)rowg * ldc + col;
          if (FLAGS & 1) ((unsigned short*)Cv)[idx] = f32_to_bf16(v);
          else           ((float*)Cv)[idx] = v;
        }
      }
    }
  }
}

// ---------------- masked softmax over S=512 keys, bf16 out ----------------
__global__ __launch_bounds__(256)
void softmax_kernel(const float* __restrict__ scores, const unsigned char* __restrict__ mask,
                    unsigned short* __restrict__ attn) {
  __shared__ float sred[8];
  const int r  = blockIdx.x;            // (h*B + b)*S + s
  const int s  = r & (S_ - 1);
  const int hb = r >> 9;
  const int b  = hb & (B_ - 1);
  const float* row = scores + (size_t)r * S_;
  const unsigned char* mrow = mask + ((size_t)b * S_ + s) * S_;
  unsigned short* arow = attn + (size_t)r * S_;
  const int t0 = threadIdx.x, t1 = threadIdx.x + 256;
  float v0 = mrow[t0] ? -1e9f : row[t0];
  float v1 = mrow[t1] ? -1e9f : row[t1];
  float mx = block_reduce_max(fmaxf(v0, v1), sred);
  float e0 = __expf(v0 - mx), e1 = __expf(v1 - mx);
  float sum = block_reduce_sum(e0 + e1, sred);
  float inv = 1.0f / sum;
  arow[t0] = f32_to_bf16(e0 * inv);
  arow[t1] = f32_to_bf16(e1 * inv);
}

// ---------------- LayerNorm over D=512, fp32 in -> bf16 out ----------------
__global__ __launch_bounds__(256)
void layernorm_bf16_kernel(const float* __restrict__ X, unsigned short* __restrict__ Y) {
  __shared__ float sred[8];
  const float* row = X + (size_t)blockIdx.x * DI_;
  unsigned short* yrow = Y + (size_t)blockIdx.x * DI_;
  float x0 = row[threadIdx.x], x1 = row[threadIdx.x + 256];
  float sum = block_reduce_sum(x0 + x1, sred);
  float sq  = block_reduce_sum(x0 * x0 + x1 * x1, sred);
  float mean = sum * (1.0f / (float)DI_);
  float var  = sq  * (1.0f / (float)DI_) - mean * mean;
  float inv  = rsqrtf(var + 1e-5f);
  yrow[threadIdx.x]       = f32_to_bf16((x0 - mean) * inv);
  yrow[threadIdx.x + 256] = f32_to_bf16((x1 - mean) * inv);
}

// ---------------- in-place log-softmax over V=30522 ----------------
__global__ __launch_bounds__(256)
void log_softmax_kernel(float* __restrict__ X, int D) {
  __shared__ float sred[8];
  float* row = X + (size_t)blockIdx.x * D;
  float mx = -3.0e38f;
  for (int c = threadIdx.x; c < D; c += 256) mx = fmaxf(mx, row[c]);
  mx = block_reduce_max(mx, sred);
  float sum = 0.0f;
  for (int c = threadIdx.x; c < D; c += 256) sum += __expf(row[c] - mx);
  sum = block_reduce_sum(sum, sred);
  float lse = mx + logf(sum);
  for (int c = threadIdx.x; c < D; c += 256) row[c] -= lse;
}

// ---------------- host orchestration ----------------
static inline void launch_cvt_t(hipStream_t st, const float* s, unsigned short* d,
                                int K, int N, int Z) {
  dim3 g(K / 32, (N + 31) / 32, Z), blk(256);
  cvt_transpose_bf16_kernel<<<g, blk, 0, st>>>(s, d, K, N);
}

template <int NCHECK, int FLAGS>
static inline void launch_gemm(hipStream_t st,
    const unsigned short* A, int lda, long long Ao, long long Ai,
    const unsigned short* Bm, int ldb, long long Bo, long long Bi,
    const float* bias, long long bo, long long bi,
    void* C, int ldc, long long Co, long long Ci,
    int M, int N, int K, int Z, int zdiv, float alpha) {
  dim3 g((N + GBN - 1) / GBN, M / GBM, Z), blk(256);
  gemm_bf16_kernel<NCHECK, FLAGS><<<g, blk, 0, st>>>(
      A, lda, Ao, Ai, Bm, ldb, Bo, Bi, bias, bo, bi, C, ldc, Co, Ci,
      M, N, K, zdiv, alpha);
}

extern "C" void kernel_launch(void* const* d_in, const int* in_sizes, int n_in,
                              void* d_out, int out_size, void* d_ws, size_t ws_size,
                              hipStream_t stream) {
  (void)in_sizes; (void)n_in; (void)out_size; (void)ws_size;

  const int*   input   = (const int*)d_in[0];
  const unsigned char* mask = (const unsigned char*)d_in[1];   // jnp.bool_
  const float* timeT   = (const float*)d_in[2];
  const int*   poiT    = (const int*)d_in[3];
  const float* s_emb   = (const float*)d_in[4];
  const float* spat    = (const float*)d_in[5];
  const float* poitab  = (const float*)d_in[6];
  const float* w_time  = (const float*)d_in[7];
  const float* b_time  = (const float*)d_in[8];
  const float* Wq = (const float*)d_in[9];  const float* bq = (const float*)d_in[10];
  const float* Wk = (const float*)d_in[11]; const float* bk = (const float*)d_in[12];
  const float* Wv = (const float*)d_in[13]; const float* bv = (const float*)d_in[14];
  const float* Wf = (const float*)d_in[15]; const float* bfp = (const float*)d_in[16];
  const float* Wo = (const float*)d_in[17]; const float* bo = (const float*)d_in[18];
  const float* W1 = (const float*)d_in[19]; const float* b1 = (const float*)d_in[20];
  const float* W2 = (const float*)d_in[21]; const float* b2 = (const float*)d_in[22];
  const float* Wtok = (const float*)d_in[23]; const float* btok = (const float*)d_in[24];
  float* out = (float*)d_out;

  // ---- workspace carve (256B aligned) ----
  char* base = (char*)d_ws; size_t off = 0;
  auto alloc = [&](size_t bytes) -> void* {
    void* p = base + off; off = (off + bytes + 255) & ~(size_t)255; return p;
  };
  unsigned short* catB    = (unsigned short*)alloc((size_t)NTOK * F_ * 2);
  unsigned short* WfT     = (unsigned short*)alloc((size_t)H_ * F_ * DI_ * 2);   // [H][DI][F]
  unsigned short* fusedB  = (unsigned short*)alloc((size_t)H_ * NTOK * DI_ * 2);
  unsigned short* WqT     = (unsigned short*)alloc((size_t)H_ * DI_ * DO_ * 2);  // [H][DO][DI]
  unsigned short* WkT     = (unsigned short*)alloc((size_t)H_ * DI_ * DO_ * 2);
  unsigned short* WvT     = (unsigned short*)alloc((size_t)H_ * DI_ * DO_ * 2);
  unsigned short* qB      = (unsigned short*)alloc((size_t)H_ * NTOK * DO_ * 2); // [H][tok][DO]
  unsigned short* kB      = (unsigned short*)alloc((size_t)H_ * NTOK * DO_ * 2); // [H][tok][DO]
  unsigned short* vT      = (unsigned short*)alloc((size_t)H_ * NTOK * DO_ * 2); // [H][DO][tok]
  float*          scoresF = (float*)alloc((size_t)H_ * B_ * S_ * S_ * 4);
  unsigned short* attnB   = (unsigned short*)alloc((size_t)H_ * B_ * S_ * S_ * 2);
  unsigned short* ctxB    = (unsigned short*)alloc((size_t)NTOK * H_ * DO_ * 2);
  unsigned short* WoT     = (unsigned short*)alloc((size_t)(H_ * DO_) * DI_ * 2); // [DI][HDO]
  float*          mhaPre  = (float*)alloc((size_t)NTOK * DI_ * 4);
  unsigned short* mhaOutB = (unsigned short*)alloc((size_t)NTOK * DI_ * 2);
  unsigned short* W1T     = (unsigned short*)alloc((size_t)DI_ * DO_ * 2);        // [DO][DI]
  unsigned short* hdnB    = (unsigned short*)alloc((size_t)NTOK * DO_ * 2);
  unsigned short* W2T     = (unsigned short*)alloc((size_t)DO_ * DI_ * 2);        // [DI][DO]
  float*          encPre  = (float*)alloc((size_t)NTOK * DI_ * 4);
  unsigned short* encB    = (unsigned short*)alloc((size_t)NTOK * DI_ * 2);
  unsigned short* WtokT   = (unsigned short*)alloc((size_t)DI_ * V_ * 2);         // [V][DI]

  // ---- weights -> bf16, transposed to [N][K] ----
  launch_cvt_t(stream, Wf,   WfT,   F_,       DI_,      H_);
  launch_cvt_t(stream, Wq,   WqT,   DI_,      DO_,      H_);
  launch_cvt_t(stream, Wk,   WkT,   DI_,      DO_,      H_);
  launch_cvt_t(stream, Wv,   WvT,   DI_,      DO_,      H_);
  launch_cvt_t(stream, Wo,   WoT,   H_ * DO_, DI_,      1);
  launch_cvt_t(stream, W1,   W1T,   DI_,      DO_,      1);
  launch_cvt_t(stream, W2,   W2T,   DO_,      DI_,      1);
  launch_cvt_t(stream, Wtok, WtokT, DI_,      V_,       1);

  // ---- embeddings + LN + concat ----
  build_cat_kernel<<<NTOK, 256, 0, stream>>>(input, poiT, timeT, s_emb, spat, poitab,
                                             w_time, b_time, catB);

  // ---- fused[h] = cat @ Wf[h] + bf[h]  (bf16 out) ----
  launch_gemm<0, 5>(stream, catB, F_, 0, 0,
                    WfT, F_, (long long)DI_ * F_, 0,
                    bfp, DI_, 0,
                    fusedB, DI_, (long long)NTOK * DI_, 0,
                    NTOK, DI_, F_, H_, 1, 1.0f);

  // ---- q/k = fused[h] @ Wq/Wk[h] + b  (row-major [tok][DO]) ----
  launch_gemm<0, 5>(stream, fusedB, DI_, (long long)NTOK * DI_, 0,
                    WqT, DI_, (long long)DO_ * DI_, 0,
                    bq, DO_, 0,
                    qB, DO_, (long long)NTOK * DO_, 0,
                    NTOK, DO_, DI_, H_, 1, 1.0f);
  launch_gemm<0, 5>(stream, fusedB, DI_, (long long)NTOK * DI_, 0,
                    WkT, DI_, (long long)DO_ * DI_, 0,
                    bk, DO_, 0,
                    kB, DO_, (long long)NTOK * DO_, 0,
                    NTOK, DO_, DI_, H_, 1, 1.0f);
  // ---- v = pure @ Wv[h] + bv, stored TRANSPOSED as vT[h][o][tok] (bias|bf16|transC) ----
  launch_gemm<0, 13>(stream, catB, F_, 0, 0,
                     WvT, DI_, (long long)DO_ * DI_, 0,
                     bv, DO_, 0,
                     vT, NTOK, (long long)DO_ * NTOK, 0,
                     NTOK, DO_, DI_, H_, 1, 1.0f);

  // ---- scores[h,b] = (q @ k^T) / sqrt(S): B = k is naturally [n=key][k=DO] ----
  launch_gemm<0, 0>(stream, qB, DO_, (long long)S_ * DO_, 0,
                    kB, DO_, (long long)S_ * DO_, 0,
                    nullptr, 0, 0,
                    scoresF, S_, (long long)S_ * S_, 0,
                    S_, S_, DO_, H_ * B_, 1, 0.04419417382415922f);

  // ---- masked softmax -> bf16 attn ----
  softmax_kernel<<<H_ * B_ * S_, 256, 0, stream>>>(scoresF, mask, attnB);

  // ---- ctx[h,b] = attn @ v: B = vT[h][o][tok], per-(h,b) base h*DO*NTOK + b*S ----
  launch_gemm<0, 1>(stream, attnB, S_, (long long)B_ * S_ * S_, (long long)S_ * S_,
                    vT, NTOK, (long long)DO_ * NTOK, (long long)S_,
                    nullptr, 0, 0,
                    ctxB, H_ * DO_, (long long)DO_, (long long)S_ * (H_ * DO_),
                    S_, DO_, S_, H_ * B_, B_, 1.0f);

  // ---- mha_out = LN(ctx @ Wo + bo) ----
  launch_gemm<0, 4>(stream, ctxB, H_ * DO_, 0, 0,
                    WoT, H_ * DO_, 0, 0,
                    bo, 0, 0,
                    mhaPre, DI_, 0, 0,
                    NTOK, DI_, H_ * DO_, 1, 1, 1.0f);
  layernorm_bf16_kernel<<<NTOK, 256, 0, stream>>>(mhaPre, mhaOutB);

  // ---- hdn = gelu(mha_out @ W1 + b1)  (bf16|gelu|bias) ----
  launch_gemm<0, 7>(stream, mhaOutB, DI_, 0, 0,
                    W1T, DI_, 0, 0,
                    b1, 0, 0,
                    hdnB, DO_, 0, 0,
                    NTOK, DO_, DI_, 1, 1, 1.0f);

  // ---- enc = LN(hdn @ W2 + b2) ----
  launch_gemm<0, 4>(stream, hdnB, DO_, 0, 0,
                    W2T, DO_, 0, 0,
                    b2, 0, 0,
                    encPre, DI_, 0, 0,
                    NTOK, DI_, DO_, 1, 1, 1.0f);
  layernorm_bf16_kernel<<<NTOK, 256, 0, stream>>>(encPre, encB);

  // ---- logits = enc @ Wtok + btok  (f32 to d_out, ragged N = 30522) ----
  launch_gemm<1, 4>(stream, encB, DI_, 0, 0,
                    WtokT, DI_, 0, 0,
                    btok, 0, 0,
                    out, V_, 0, 0,
                    NTOK, V_, DI_, 1, 1, 1.0f);

  // ---- log-softmax in place ----
  log_softmax_kernel<<<NTOK, 256, 0, stream>>>(out, V_);
}
